// MultiHeadAttention_32968168964551
// MI455X (gfx1250) — matrix operand order repaired and checked by reading, not compiled, not run on hardware
//
#include <hip/hip_runtime.h>

#define BATCH 2
#define SEQ   2048
#define DM    1024
#define NH    16
#define DK    64
#define MTOT  (BATCH*SEQ)   // 4096
#define NQKV  (3*DM)        // 3072

typedef __attribute__((ext_vector_type(16))) __bf16 v16bf;
typedef __attribute__((ext_vector_type(8)))  __bf16 v8bf;
typedef __attribute__((ext_vector_type(8)))  float  v8f;
typedef __attribute__((ext_vector_type(4)))  unsigned int v4u;
typedef __attribute__((ext_vector_type(8)))  int v8i_;
typedef __attribute__((ext_vector_type(4)))  int v4i_;

static __device__ __forceinline__ v16bf cat16(v8bf lo, v8bf hi) {
  return __builtin_shufflevector(lo, hi, 0,1,2,3,4,5,6,7,8,9,10,11,12,13,14,15);
}
static __device__ __forceinline__ v16bf frag16(const __bf16* p0, const __bf16* p1) {
  return cat16(*(const v8bf*)p0, *(const v8bf*)p1);
}
static __device__ __forceinline__ v8f wmma_bf16(v16bf a, v16bf b, v8f c) {
  return __builtin_amdgcn_wmma_f32_16x16x32_bf16(false, a, false, b, (short)0, c,
                                                 false, false);
}

// ---- Tensor Data Mover: 2D bf16 tile (tile_d0 contiguous elems x tile_d1 rows)
// from global (row stride = stride0 elements) into LDS at byte offset lds_off.
// D# layout per CDNA5 ISA ch.8: group0 = count/lds/global/type, group1 = dims.
// This toolchain exposes the 6-arg builtin: (g0, g1, g2, g3, g_extra, cpol).
static __device__ __forceinline__ void tdm_load_2d(const __bf16* gptr,
                                                   unsigned lds_off,
                                                   int tensor_d0, int tensor_d1,
                                                   int tile_d0, int tile_d1,
                                                   int stride0) {
  unsigned long long ga = (unsigned long long)(uintptr_t)gptr;
  v4u g0;
  g0.x = 1u;                                   // count=1 (valid descriptor)
  g0.y = lds_off;                              // lds_addr (bytes)
  g0.z = (unsigned)(ga & 0xFFFFFFFFu);         // global_addr[31:0]
  g0.w = (unsigned)((ga >> 32) & 0x1FFFFFFu)   // global_addr[56:32]
         | (2u << 30);                         // type=2 ("image")
  v8i_ g1;
  g1[0] = (1 << 16);                           // data_size=1 -> 2 bytes/elem
  g1[1] = (tensor_d0 & 0xFFFF) << 16;          // tensor_dim0[15:0]
  g1[2] = ((tensor_d0 >> 16) & 0xFFFF) | ((tensor_d1 & 0xFFFF) << 16);
  g1[3] = ((tensor_d1 >> 16) & 0xFFFF) | ((tile_d0 & 0xFFFF) << 16);
  g1[4] = (tile_d1 & 0xFFFF);                  // tile_dim1; tile_dim2=0
  g1[5] = stride0;                             // tensor_dim0_stride[31:0]
  g1[6] = 0;                                   // stride0 hi / dim1_stride lo
  g1[7] = 0;
  v4i_ z4 = {0, 0, 0, 0};
  v8i_ z8 = {0, 0, 0, 0, 0, 0, 0, 0};
  __builtin_amdgcn_tensor_load_to_lds(g0, g1, z4, z4, z8, 0);
}

// ---------- f32 -> bf16 convert ----------
__global__ __launch_bounds__(256) void cvt_f32_bf16(const float* __restrict__ s,
                                                    __bf16* __restrict__ d, int n) {
  int i = blockIdx.x * blockDim.x + threadIdx.x;
  if (i < n) d[i] = (__bf16)s[i];
}

// ---------- f32 [K x N] -> bf16 transposed [N x K] ----------
__global__ __launch_bounds__(256) void transpose_f32_bf16(const float* __restrict__ s,
                                                          __bf16* __restrict__ d,
                                                          int K, int N) {
  int i = blockIdx.x * blockDim.x + threadIdx.x;
  if (i < K * N) {
    int n = i / K;
    int k = i - n * K;
    d[i] = (__bf16)s[(size_t)k * N + n];
  }
}

// ---------- QKV projection GEMM: [4096x1024] @ [1024x3072] + bias ----------
// Block = 128 rows x 64 cols (8 waves x 16 rows). B panel tiles (64x32) are
// DMA'd into double-buffered LDS by the TDM (wave 0), all waves read B
// fragments with ds_load_b128; A fragments are direct global_load_b128.
__global__ __launch_bounds__(256) void qkv_gemm_kernel(
    const __bf16* __restrict__ A,    // xb   [MTOT x DM]
    const __bf16* __restrict__ BT,   // wqkv^T [NQKV x DM]
    const float*  __restrict__ bias, // [NQKV]
    __bf16* __restrict__ Qb, __bf16* __restrict__ Kb, __bf16* __restrict__ Vt) {
  __shared__ __align__(16) __bf16 bsm[2][64][32];  // 8 KiB double buffer
  const int NSTRIP = NQKV / 64;  // 48
  int w = threadIdx.x >> 5;
  int lane = threadIdx.x & 31;
  int mtb = blockIdx.x / NSTRIP;
  int ns  = blockIdx.x - mtb * NSTRIP;
  int row0 = mtb * 128 + w * 16;
  int col0 = ns * 64;
  int ln = lane & 15, hi = lane >> 4;
  int ab = hi * 16, bb = hi * 16;

  const __bf16* arow = A + (size_t)(row0 + ln) * DM;
  const __bf16* bpanel = BT + (size_t)col0 * DM;   // 64 x 1024 panel

  if (threadIdx.x < 32)
    tdm_load_2d(bpanel, (unsigned)(uintptr_t)&bsm[0][0][0], DM, 64, 32, 64, DM);

  v8f acc[4] = {};
  for (int s = 0; s < DM / 32; ++s) {
    int k0 = s * 32;
    int buf = s & 1;
    if (threadIdx.x < 32) __builtin_amdgcn_s_wait_tensorcnt(0);
    __syncthreads();  // publish bsm[buf]; everyone done reading bsm[buf^1]
    if ((k0 + 32 < DM) && threadIdx.x < 32)
      tdm_load_2d(bpanel + (k0 + 32), (unsigned)(uintptr_t)&bsm[buf ^ 1][0][0],
                  DM, 64, 32, 64, DM);
    v16bf a = frag16(arow + k0 + ab, arow + k0 + 8 + ab);
#pragma unroll
    for (int t = 0; t < 4; ++t) {
      const __bf16* bp = &bsm[buf][t * 16 + ln][bb];
      acc[t] = wmma_bf16(a, frag16(bp, bp + 8), acc[t]);
    }
    __syncthreads();  // done reading bsm[buf] before it is overwritten
  }
#pragma unroll
  for (int t = 0; t < 4; ++t) {
    int j = col0 + t * 16 + ln;
    float bj = bias[j];
    int which = j >> 10;           // 0=Q 1=K 2=V
    int h = (j & (DM - 1)) >> 6;
    int d = j & 63;
#pragma unroll
    for (int r = 0; r < 8; ++r) {
      int i = row0 + hi * 8 + r;   // C-layout row
      int b_ = i >> 11;
      int sq = i & (SEQ - 1);
      __bf16 val = (__bf16)(acc[t][r] + bj);
      if (which == 0)
        Qb[(((size_t)(b_ * NH + h)) * SEQ + sq) * DK + d] = val;
      else if (which == 1)
        Kb[(((size_t)(b_ * NH + h)) * SEQ + sq) * DK + d] = val;
      else
        Vt[(((size_t)(b_ * NH + h)) * DK + d) * SEQ + sq] = val;
    }
  }
}

// ---------- Flash-style causal attention ----------
__global__ __launch_bounds__(256) void attn_kernel(
    const __bf16* __restrict__ Qb, const __bf16* __restrict__ Kb,
    const __bf16* __restrict__ Vt, __bf16* __restrict__ O /* [B,S,H*DK] */) {
  __shared__ __align__(16) __bf16 plds[8][16][32];  // per-wave P staging
  int w = threadIdx.x >> 5;
  int lane = threadIdx.x & 31;
  int gw = blockIdx.x * 8 + w;
  int qt = gw & 127;        // 128 query tiles of 16
  int bh = gw >> 7;         // 0..31
  int b_ = bh >> 4, h = bh & 15;
  const __bf16* Q = Qb + (size_t)bh * SEQ * DK;
  const __bf16* K = Kb + (size_t)bh * SEQ * DK;
  const __bf16* V = Vt + (size_t)bh * DK * SEQ;
  int ln = lane & 15, hi = lane >> 4;
  int ab = hi * 16, bb = hi * 16;

  v16bf qa[2];
  const __bf16* qrow = Q + (size_t)(qt * 16 + ln) * DK;
#pragma unroll
  for (int c = 0; c < 2; ++c)
    qa[c] = frag16(qrow + c * 32 + ab, qrow + c * 32 + 8 + ab);

  v8f o[4] = {};
  float rmax[8], rsum[8];
#pragma unroll
  for (int r = 0; r < 8; ++r) { rmax[r] = -__builtin_inff(); rsum[r] = 0.f; }
  const float scale = 0.125f;  // 1/sqrt(64)

  for (int kt = 0; kt <= qt; kt += 2) {
    int key0 = kt * 16;
    bool t1v = (kt + 1) <= qt;
    v8f s0 = {}, s1 = {};
#pragma unroll
    for (int c = 0; c < 2; ++c) {
      const __bf16* kp = K + (size_t)(key0 + ln) * DK + c * 32 + bb;
      s0 = wmma_bf16(qa[c], frag16(kp, kp + 8), s0);
    }
    if (t1v) {
#pragma unroll
      for (int c = 0; c < 2; ++c) {
        const __bf16* kp = K + (size_t)(key0 + 16 + ln) * DK + c * 32 + bb;
        s1 = wmma_bf16(qa[c], frag16(kp, kp + 8), s1);
      }
    }
#pragma unroll
    for (int r = 0; r < 8; ++r) {
      int qg = qt * 16 + hi * 8 + r;
      float x0 = (key0 + ln <= qg) ? s0[r] * scale : -__builtin_inff();
      float x1 = (t1v && (key0 + 16 + ln <= qg)) ? s1[r] * scale
                                                 : -__builtin_inff();
      float tmax = fmaxf(x0, x1);
#pragma unroll
      for (int off = 1; off < 16; off <<= 1)
        tmax = fmaxf(tmax, __shfl_xor(tmax, off, 32));
      float nm = fmaxf(rmax[r], tmax);
      float corr = __expf(rmax[r] - nm);
      rmax[r] = nm;
      float p0 = __expf(x0 - nm);
      float p1 = __expf(x1 - nm);
      float ps = p0 + p1;
#pragma unroll
      for (int off = 1; off < 16; off <<= 1) ps += __shfl_xor(ps, off, 32);
      rsum[r] = rsum[r] * corr + ps;
      o[0][r] *= corr; o[1][r] *= corr; o[2][r] *= corr; o[3][r] *= corr;
      plds[w][hi * 8 + r][ln]      = (__bf16)p0;
      plds[w][hi * 8 + r][16 + ln] = (__bf16)p1;
    }
    // intra-wave LDS handoff; DS ops are in-order per wave, so only a DS-count
    // wait is needed (waves in this block have different trip counts -> no
    // block barrier allowed here).
    asm volatile("s_wait_dscnt 0" ::: "memory");
    const __bf16* pr = &plds[w][ln][0];
    v16bf pa = frag16(pr + ab, pr + 8 + ab);
#pragma unroll
    for (int t = 0; t < 4; ++t) {
      const __bf16* vp = V + (size_t)(t * 16 + ln) * SEQ + key0 + bb;
      o[t] = wmma_bf16(pa, frag16(vp, vp + 8), o[t]);
    }
  }
#pragma unroll
  for (int r = 0; r < 8; ++r) rsum[r] = 1.f / rsum[r];
#pragma unroll
  for (int t = 0; t < 4; ++t)
#pragma unroll
    for (int r = 0; r < 8; ++r) {
      int srow = qt * 16 + hi * 8 + r;
      size_t idx = ((size_t)(b_ * SEQ + srow)) * DM + h * DK + t * 16 + ln;
      O[idx] = (__bf16)(o[t][r] * rsum[r]);
    }
}

// ---------- output projection GEMM (same TDM scheme, N=1024, f32 out) ----------
__global__ __launch_bounds__(256) void out_gemm_kernel(
    const __bf16* __restrict__ A,    // attn out [MTOT x DM]
    const __bf16* __restrict__ BT,   // w_out^T [DM x DM]
    const float*  __restrict__ bias, // [DM]
    float* __restrict__ out) {
  __shared__ __align__(16) __bf16 bsm[2][64][32];
  const int NSTRIP = DM / 64;  // 16
  int w = threadIdx.x >> 5;
  int lane = threadIdx.x & 31;
  int mtb = blockIdx.x / NSTRIP;
  int ns  = blockIdx.x - mtb * NSTRIP;
  int row0 = mtb * 128 + w * 16;
  int col0 = ns * 64;
  int ln = lane & 15, hi = lane >> 4;
  int ab = hi * 16, bb = hi * 16;

  const __bf16* arow = A + (size_t)(row0 + ln) * DM;
  const __bf16* bpanel = BT + (size_t)col0 * DM;

  if (threadIdx.x < 32)
    tdm_load_2d(bpanel, (unsigned)(uintptr_t)&bsm[0][0][0], DM, 64, 32, 64, DM);

  v8f acc[4] = {};
  for (int s = 0; s < DM / 32; ++s) {
    int k0 = s * 32;
    int buf = s & 1;
    if (threadIdx.x < 32) __builtin_amdgcn_s_wait_tensorcnt(0);
    __syncthreads();
    if ((k0 + 32 < DM) && threadIdx.x < 32)
      tdm_load_2d(bpanel + (k0 + 32), (unsigned)(uintptr_t)&bsm[buf ^ 1][0][0],
                  DM, 64, 32, 64, DM);
    v16bf a = frag16(arow + k0 + ab, arow + k0 + 8 + ab);
#pragma unroll
    for (int t = 0; t < 4; ++t) {
      const __bf16* bp = &bsm[buf][t * 16 + ln][bb];
      acc[t] = wmma_bf16(a, frag16(bp, bp + 8), acc[t]);
    }
    __syncthreads();
  }
#pragma unroll
  for (int t = 0; t < 4; ++t) {
    int j = col0 + t * 16 + ln;
    float bj = bias[j];
#pragma unroll
    for (int r = 0; r < 8; ++r) {
      int i = row0 + hi * 8 + r;
      out[(size_t)i * DM + j] = acc[t][r] + bj;
    }
  }
}

extern "C" void kernel_launch(void* const* d_in, const int* in_sizes, int n_in,
                              void* d_out, int out_size, void* d_ws, size_t ws_size,
                              hipStream_t stream) {
  const float* x     = (const float*)d_in[0];
  const float* w_qkv = (const float*)d_in[1];
  const float* b_qkv = (const float*)d_in[2];
  const float* w_out = (const float*)d_in[3];
  const float* b_out = (const float*)d_in[4];
  float* out = (float*)d_out;

  char* ws = (char*)d_ws;
  size_t off = 0;
  __bf16* xb  = (__bf16*)(ws + off); off += (size_t)MTOT * DM * 2;   // 8 MiB
  __bf16* wqT = (__bf16*)(ws + off); off += (size_t)NQKV * DM * 2;   // 6 MiB
  __bf16* woT = (__bf16*)(ws + off); off += (size_t)DM * DM * 2;     // 2 MiB
  __bf16* Qb  = (__bf16*)(ws + off); off += (size_t)MTOT * DM * 2;   // 8 MiB
  __bf16* Kb  = (__bf16*)(ws + off); off += (size_t)MTOT * DM * 2;   // 8 MiB
  __bf16* Vt  = (__bf16*)(ws + off); off += (size_t)MTOT * DM * 2;   // 8 MiB
  __bf16* AO  = (__bf16*)(ws + off); off += (size_t)MTOT * DM * 2;   // 8 MiB

  cvt_f32_bf16<<<(MTOT * DM) / 256, 256, 0, stream>>>(x, xb, MTOT * DM);
  transpose_f32_bf16<<<(NQKV * DM) / 256, 256, 0, stream>>>(w_qkv, wqT, DM, NQKV);
  transpose_f32_bf16<<<(DM * DM) / 256, 256, 0, stream>>>(w_out, woT, DM, DM);

  // 32 row-blocks (128 rows) * 48 col-strips (64 cols) = 1536 blocks
  qkv_gemm_kernel<<<(MTOT / 128) * (NQKV / 64), 256, 0, stream>>>(
      xb, wqT, b_qkv, Qb, Kb, Vt);

  // 2*16*128 = 4096 waves, 8 per block
  attn_kernel<<<(BATCH * NH * (SEQ / 16)) / 8, 256, 0, stream>>>(Qb, Kb, Vt, AO);

  // 32 * 16 = 512 blocks
  out_gemm_kernel<<<(MTOT / 128) * (DM / 64), 256, 0, stream>>>(
      AO, woT, b_out, out);
}
